// SimpleBertModel_49675591745835
// MI455X (gfx1250) — matrix operand order repaired and checked by reading, not compiled, not run on hardware
//
#include <hip/hip_runtime.h>
#include <stdint.h>

// ---------------------------------------------------------------------------
// CDNA5 / gfx1250: wave32, WMMA 16x16x32 bf16 -> f32 accumulate.
// ---------------------------------------------------------------------------
typedef __bf16 bf16_t;
typedef __attribute__((ext_vector_type(16))) __bf16 v16bf;
typedef __attribute__((ext_vector_type(8)))  __bf16 v8bf;
typedef __attribute__((ext_vector_type(8)))  float  v8f;

#define WAVE 32
#define NT 4          // N-subtiles (16 wide each) per wave -> 16x64 C per wave

static __device__ __forceinline__ bf16_t u16_as_bf16(unsigned short v)
{
    return __builtin_bit_cast(bf16_t, v);
}

// ---------------------------------------------------------------------------
// Generic batched GEMM:
//   C[z] = act( (alpha * A[z] @ op(B[z]) + bias + residual) / rowDiv )
// A: bf16 row-major [M,K] (lda), ALWAYS 16B-aligned with lda%8==0 (host
// guarantees). B: BTRANS ? row-major [N,K] holding B^T (vectorizable along K)
//                        : [K,N] (lane-contiguous scalar loads).
// One wave computes a 16x64 tile; 4 independent WMMAs per K-step share one
// A fragment. Batch z -> (b=z/Hn, h=z%Hn) with per-operand strides.
//
// FULLTILE=true (M%16==0, N%64==0, K%32==0): no edge handling compiled in.
// FULLTILE=false: edge tiles / K-tails use clamped UNCONDITIONAL u16 loads
// masked by a bitwise AND ("raw & msk" uses the loaded value unconditionally,
// so LLVM cannot sink the load into exec-masked blocks).
// ---------------------------------------------------------------------------
template <bool FULLTILE, bool BTRANS>
__global__ __launch_bounds__(WAVE)
void gemm_bf16_wmma(const bf16_t* __restrict__ A, const bf16_t* __restrict__ Bm,
                    float* __restrict__ Cf, bf16_t* __restrict__ Cb,
                    const float* __restrict__ bias, const float* __restrict__ residual,
                    const float* __restrict__ rowDiv, int applyRelu,
                    int M, int N, int K,
                    long lda, long ldb, long ldc,
                    float alpha,
                    int Hn, long sAb, long sAh, long sBb, long sBh, long sCb, long sCh)
{
    const int z  = blockIdx.z;
    const int bb = z / Hn;
    const int hh = z - bb * Hn;
    const bf16_t* __restrict__ Ab = A  + (long)bb * sAb + (long)hh * sAh;
    const bf16_t* __restrict__ Bb = Bm + (long)bb * sBb + (long)hh * sBh;
    const long cOff = (long)bb * sCb + (long)hh * sCh;

    const int lane   = threadIdx.x & (WAVE - 1);
    const int m16    = lane & 15;
    const int hiHalf = lane >> 4;           // 0: lanes 0-15, 1: lanes 16-31
    const int tileM  = blockIdx.y;
    const int tileN0 = blockIdx.x * NT;

    const int rowA = tileM * 16 + m16;      // A row held by this lane
    const int kbA  = hiHalf * 16;           // A 16x32: lanes 0-15 K=0..15, 16-31 K=16..31
    const int kbB  = hiHalf * 16;           // B Kx16: lanes 0-15 K=0..15, 16-31 K=16..31

    v8f acc[NT] = {};

    // Unguarded K-step: single compile-time-selected load path.
    auto fast_step = [&](int k0) {
        const bf16_t* pa = Ab + (long)rowA * lda + k0 + kbA;
        __builtin_prefetch(pa + 256, 0, 1);               // global_prefetch_b8
        v8bf a0 = *reinterpret_cast<const v8bf*>(pa);
        v8bf a1 = *reinterpret_cast<const v8bf*>(pa + 8);
        v16bf afr = __builtin_shufflevector(a0, a1, 0, 1, 2, 3, 4, 5, 6, 7,
                                            8, 9, 10, 11, 12, 13, 14, 15);
#pragma unroll
        for (int nt = 0; nt < NT; ++nt) {
            const int colB = (tileN0 + nt) * 16 + m16;
            v16bf bfr;
            if (BTRANS) {
                const bf16_t* pb = Bb + (long)colB * ldb + k0 + kbB;
                if (nt == 0) __builtin_prefetch(pb + 256, 0, 1);
                v8bf b0 = *reinterpret_cast<const v8bf*>(pb);
                v8bf b1 = *reinterpret_cast<const v8bf*>(pb + 8);
                bfr = __builtin_shufflevector(b0, b1, 0, 1, 2, 3, 4, 5, 6, 7,
                                              8, 9, 10, 11, 12, 13, 14, 15);
            } else {
                // lane-contiguous across the half-wave -> coalesced u16 loads
#pragma unroll
                for (int q = 0; q < 16; ++q)
                    bfr[q] = Bb[(long)(k0 + kbB + q) * ldb + colB];
            }
            acc[nt] = __builtin_amdgcn_wmma_f32_16x16x32_bf16(false, afr, false, bfr,
                                                              (short)0, acc[nt], false, false);
        }
    };

    if (FULLTILE) {
        for (int k0 = 0; k0 < K; k0 += 32) fast_step(k0);
    } else {
        const unsigned short* __restrict__ Au = (const unsigned short*)Ab;
        const unsigned short* __restrict__ Bu = (const unsigned short*)Bb;

        // Guarded K-step: clamped unconditional loads + integer masking.
        auto guarded_step = [&](int k0) {
            const bool rowOK = rowA < M;
            const long aRow  = (long)(rowOK ? rowA : (M - 1)) * lda;
            const unsigned short mRow = rowOK ? (unsigned short)0xFFFFu : (unsigned short)0;
            v16bf afr;
#pragma unroll
            for (int j = 0; j < 8; ++j) {
                const int kA = kbA + (j << 1);
#pragma unroll
                for (int h2 = 0; h2 < 2; ++h2) {
                    const int ka = k0 + kA + h2;
                    const int kc = (ka < K) ? ka : (K - 1);
                    const unsigned short raw = Au[aRow + kc];        // always in-bounds
                    const unsigned short msk = (ka < K) ? mRow : (unsigned short)0;
                    afr[2 * j + h2] = u16_as_bf16((unsigned short)(raw & msk));
                }
            }
#pragma unroll
            for (int nt = 0; nt < NT; ++nt) {
                const int  colB  = (tileN0 + nt) * 16 + m16;
                const bool colOK = colB < N;
                const int  colC  = colOK ? colB : (N - 1);
                const unsigned short mCol = colOK ? (unsigned short)0xFFFFu : (unsigned short)0;
                v16bf bfr;
#pragma unroll
                for (int j = 0; j < 8; ++j) {
#pragma unroll
                    for (int h2 = 0; h2 < 2; ++h2) {
                        const int kb = k0 + kbB + (j << 1) + h2;
                        const int kc = (kb < K) ? kb : (K - 1);
                        unsigned short raw;
                        if (BTRANS) raw = Bu[(long)colC * ldb + kc];
                        else        raw = Bu[(long)kc * ldb + colC];
                        const unsigned short msk = (kb < K) ? mCol : (unsigned short)0;
                        bfr[2 * j + h2] = u16_as_bf16((unsigned short)(raw & msk));
                    }
                }
                acc[nt] = __builtin_amdgcn_wmma_f32_16x16x32_bf16(false, afr, false, bfr,
                                                                  (short)0, acc[nt], false, false);
            }
        };

        const bool fullM = (tileM * 16 + 16) <= M;
        const bool fullN = (tileN0 * 16 + NT * 16) <= N;
        const int  Kfull = K & ~31;
        if (fullM && fullN) {
            for (int k0 = 0; k0 < Kfull; k0 += 32) fast_step(k0);
            if (Kfull < K) guarded_step(Kfull);
        } else {
            for (int k0 = 0; k0 < K; k0 += 32) guarded_step(k0);
        }
    }

    // ---- epilogue ----
#pragma unroll
    for (int nt = 0; nt < NT; ++nt) {
        const int col = (tileN0 + nt) * 16 + m16;
#pragma unroll
        for (int r = 0; r < 8; ++r) {
            const int orow = tileM * 16 + r + hiHalf * 8;   // f32 C/D lane layout
            if (orow < M && col < N) {
                float v = acc[nt][r] * alpha;
                if (bias)     v += bias[col];
                const long idx = cOff + (long)orow * ldc + col;
                if (residual) v += residual[idx];
                if (rowDiv)   v /= rowDiv[orow];
                if (applyRelu) v = v > 0.0f ? v : 0.0f;
                if (Cf) Cf[idx] = v;
                if (Cb) Cb[idx] = (bf16_t)v;
            }
        }
    }
}

// ---------------------------------------------------------------------------
// Elementwise / reduction helpers (bandwidth-trivial; f32 VALU)
// ---------------------------------------------------------------------------
__global__ void cast_f32_to_bf16_k(const float* __restrict__ in, bf16_t* __restrict__ out, long n)
{
    long i = (long)blockIdx.x * blockDim.x + threadIdx.x;
    if (i < n) out[i] = (bf16_t)in[i];
}

// out[c][r] = (bf16) in[r][c] ; LDS tile for coalescing on both sides
__global__ __launch_bounds__(256)
void cast_transpose_k(const float* __restrict__ in, bf16_t* __restrict__ out, int R, int C)
{
    __shared__ float tile[16][17];
    const int tx = threadIdx.x & 15, ty = threadIdx.x >> 4;
    const int r = blockIdx.y * 16 + ty, c = blockIdx.x * 16 + tx;
    if (r < R && c < C) tile[ty][tx] = in[(long)r * C + c];
    __syncthreads();
    const int orow = blockIdx.x * 16 + ty;   // c
    const int ocol = blockIdx.y * 16 + tx;   // r
    if (orow < C && ocol < R) out[(long)orow * R + ocol] = (bf16_t)tile[tx][ty];
}

__global__ __launch_bounds__(256)
void layernorm_rows_k(const float* __restrict__ x, const float* __restrict__ g,
                      const float* __restrict__ b, float* __restrict__ outF,
                      bf16_t* __restrict__ outB, long ldOut, int D)
{
    const int r = blockIdx.x;
    const float* xr = x + (long)r * D;
    __shared__ float red[256];
    const int t = threadIdx.x;
    float s = 0.f, s2 = 0.f;
    for (int i = t; i < D; i += 256) { float v = xr[i]; s += v; s2 += v * v; }
    red[t] = s; __syncthreads();
    for (int o = 128; o > 0; o >>= 1) { if (t < o) red[t] += red[t + o]; __syncthreads(); }
    const float mean = red[0] / (float)D; __syncthreads();
    red[t] = s2; __syncthreads();
    for (int o = 128; o > 0; o >>= 1) { if (t < o) red[t] += red[t + o]; __syncthreads(); }
    const float var = red[0] / (float)D - mean * mean;
    const float rs = rsqrtf(var + 1e-5f);
    for (int i = t; i < D; i += 256) {
        float v = (xr[i] - mean) * rs * g[i] + b[i];
        if (outF) outF[(long)r * ldOut + i] = v;
        if (outB) outB[(long)r * ldOut + i] = (bf16_t)v;
    }
}

// softmax over first W of Wpad columns; pad columns get exact zeros so a
// padded-K GEMM over Wpad contributes nothing.
__global__ __launch_bounds__(WAVE)
void softmax_rows_k(const float* __restrict__ x, bf16_t* __restrict__ y, int W, int Wpad)
{
    const long r = blockIdx.x;
    const float* xr = x + r * Wpad;
    bf16_t* yr = y + r * Wpad;
    const int t = threadIdx.x;
    float mx = -3.0e38f;
    for (int i = t; i < W; i += WAVE) mx = fmaxf(mx, xr[i]);
#pragma unroll
    for (int o = 16; o > 0; o >>= 1) mx = fmaxf(mx, __shfl_xor(mx, o, WAVE));
    float s = 0.f;
    for (int i = t; i < W; i += WAVE) s += __expf(xr[i] - mx);
#pragma unroll
    for (int o = 16; o > 0; o >>= 1) s += __shfl_xor(s, o, WAVE);
    const float inv = 1.0f / s;
    for (int i = t; i < W; i += WAVE) yr[i] = (bf16_t)(__expf(xr[i] - mx) * inv);
    for (int i = W + t; i < Wpad; i += WAVE) yr[i] = (bf16_t)0.0f;
}

__global__ __launch_bounds__(256)
void rownorm_k(const float* __restrict__ x, float* __restrict__ nrm, int D)
{
    const int r = blockIdx.x;
    const float* xr = x + (long)r * D;
    __shared__ float red[256];
    const int t = threadIdx.x;
    float s = 0.f;
    for (int i = t; i < D; i += 256) { float v = xr[i]; s += v * v; }
    red[t] = s; __syncthreads();
    for (int o = 128; o > 0; o >>= 1) { if (t < o) red[t] += red[t + o]; __syncthreads(); }
    if (t == 0) nrm[r] = sqrtf(red[0]);
}

__global__ __launch_bounds__(256)
void sim_adj_k(const float* __restrict__ dot, const float* __restrict__ nrm,
               const float* __restrict__ adj, bf16_t* __restrict__ adj_bf,
               bf16_t* __restrict__ adjti_bf, float* __restrict__ den,
               float* __restrict__ denti, int Ln)
{
    const int b = blockIdx.y, l = blockIdx.x;
    const float nl = nrm[b * Ln + l];
    const long base = ((long)b * Ln + l) * Ln;
    __shared__ float r1[256], r2[256];
    const int t = threadIdx.x;
    float sa = 0.f, sat = 0.f;
    for (int m = t; m < Ln; m += 256) {
        const float a  = adj[base + m];
        const float nm = nrm[b * Ln + m];
        const float sim = dot[base + m] / fmaxf(nl * nm, 1e-16f);
        const float at = a * sim;
        adj_bf[base + m]   = (bf16_t)a;
        adjti_bf[base + m] = (bf16_t)at;
        sa += a; sat += at;
    }
    r1[t] = sa; r2[t] = sat; __syncthreads();
    for (int o = 128; o > 0; o >>= 1) { if (t < o) { r1[t] += r1[t + o]; r2[t] += r2[t + o]; } __syncthreads(); }
    if (t == 0) { den[b * Ln + l] = r1[0] + 1.0f; denti[b * Ln + l] = r2[0] + 1.0f; }
}

__global__ __launch_bounds__(256)
void span_pool_k(const float* __restrict__ x, const int* __restrict__ tran,
                 const int* __restrict__ wlen, bf16_t* __restrict__ outb,
                 int Ln, int Dn)
{
    const int b = blockIdx.y, t = blockIdx.x;   // t in [0, Ln)
    const long obase = ((long)b * Ln + t) * Dn;
    if (t == 0) {
        for (int d = threadIdx.x; d < Dn; d += 256) outb[obase + d] = (bf16_t)0.0f;
        return;
    }
    const int S = Ln - 1, ti = t - 1;
    const int st = tran[((long)b * S + ti) * 2 + 0];
    const int en = tran[((long)b * S + ti) * 2 + 1];
    const bool ok = ti < wlen[b];
    for (int d = threadIdx.x; d < Dn; d += 256) {
        float s = 0.f;
        if (ok)
            for (int si = st; si < en; ++si)
                s += x[((long)b * Ln + 1 + si) * Dn + d];
        outb[obase + d] = (bf16_t)s;
    }
}

__global__ __launch_bounds__(256)
void masked_mean_k(const float* __restrict__ g, const float* __restrict__ mask,
                   bf16_t* __restrict__ out, long ldOut, long colOff, int Ln, int Dn)
{
    const int b = blockIdx.x;
    float wn = 0.f;
    for (int l = 0; l < Ln; ++l) wn += mask[b * Ln + l];
    for (int d = threadIdx.x; d < Dn; d += 256) {
        float s = 0.f;
        for (int l = 0; l < Ln; ++l) {
            const float mk = mask[b * Ln + l];
            if (mk != 0.f) s += g[((long)b * Ln + l) * Dn + d] * mk;
        }
        out[(long)b * ldOut + colOff + d] = (bf16_t)(s / wn);
    }
}

// ---------------------------------------------------------------------------
// Host-side orchestration
// ---------------------------------------------------------------------------
static inline int ceil_div(int a, int b) { return (a + b - 1) / b; }

extern "C" void kernel_launch(void* const* d_in, const int* in_sizes, int n_in,
                              void* d_out, int out_size, void* d_ws, size_t ws_size,
                              hipStream_t stream)
{
    (void)in_sizes; (void)n_in; (void)out_size; (void)ws_size;

    constexpr int B = 64, L = 160, NV = 197, D = 768, H = 4, HD = 3072, NC = 3;
    constexpr int NVP = 256;                 // NV padded to a multiple of 64 (N-tile)
    constexpr int ML = B * L, BH = B * H;

    const float* text_hidden = (const float*)d_in[0];
    const float* vit_feature = (const float*)d_in[1];
    const float* pooler_out  = (const float*)d_in[2];
    const float* target_mask = (const float*)d_in[3];
    const float* adj         = (const float*)d_in[4];
    const float* ln_g        = (const float*)d_in[5];
    const float* ln_b        = (const float*)d_in[6];
    const float* wq          = (const float*)d_in[7];
    const float* wk          = (const float*)d_in[8];
    const float* wv          = (const float*)d_in[9];
    const float* fc_w        = (const float*)d_in[10];
    const float* mha_ln_g    = (const float*)d_in[11];
    const float* mha_ln_b    = (const float*)d_in[12];
    const float* w1b         = (const float*)d_in[15];
    const float* b1b         = (const float*)d_in[16];
    const float* w2b         = (const float*)d_in[19];
    const float* b2b         = (const float*)d_in[20];
    const float* loc_w       = (const float*)d_in[21];
    const float* loc_b       = (const float*)d_in[22];
    const float* out_w       = (const float*)d_in[23];
    const float* out_b       = (const float*)d_in[24];
    const int*   word_length = (const int*)d_in[25];
    const int*   tran        = (const int*)d_in[26];
    float* out = (float*)d_out;

    // ---- workspace bump allocator (256B aligned chunks) ----
    size_t off = 0;
    auto alloc = [&](size_t bytes) -> char* {
        char* p = (char*)d_ws + off;
        off += (bytes + 255) & ~(size_t)255;
        return p;
    };
    float*  tf32     = (float*) alloc((size_t)ML * D * 4);    // LN(text_hidden)
    bf16_t* tfbf     = (bf16_t*)alloc((size_t)ML * D * 2);
    bf16_t* vitbf    = (bf16_t*)alloc((size_t)B * NV * D * 2);
    bf16_t* wqT      = (bf16_t*)alloc((size_t)D * HD * 2);    // [HD][D] = wq^T
    bf16_t* wkT      = (bf16_t*)alloc((size_t)D * HD * 2);
    bf16_t* wvT      = (bf16_t*)alloc((size_t)D * HD * 2);
    bf16_t* fcwT     = (bf16_t*)alloc((size_t)HD * D * 2);    // [D][HD]
    bf16_t* w1bT     = (bf16_t*)alloc((size_t)D * D * 2);
    bf16_t* w2bT     = (bf16_t*)alloc((size_t)D * D * 2);
    bf16_t* locwT    = (bf16_t*)alloc((size_t)2 * D * D * 2);  // [D][2D]
    bf16_t* outwT    = (bf16_t*)alloc((size_t)2 * D * NC * 2); // [NC][2D]
    bf16_t* q_bf     = (bf16_t*)alloc((size_t)ML * HD * 2);
    bf16_t* k_bf     = (bf16_t*)alloc((size_t)B * NVP * HD * 2);   // padded rows
    bf16_t* v_bf     = (bf16_t*)alloc((size_t)B * NVP * HD * 2);   // padded rows
    float*  attnf    = (float*) alloc((size_t)BH * L * NVP * 4);   // padded cols
    bf16_t* probs_bf = (bf16_t*)alloc((size_t)BH * L * NVP * 2);   // padded cols (zeroed)
    bf16_t* ctx_bf   = (bf16_t*)alloc((size_t)ML * HD * 2);
    float*  pre_f    = (float*) alloc((size_t)ML * D * 4);    // ctx@fc_w + text_feat
    float*  tii_f    = (float*) alloc((size_t)ML * D * 4);    // LN(pre)
    bf16_t* tii_bf   = (bf16_t*)alloc((size_t)ML * D * 2);
    float*  dot_f    = (float*) alloc((size_t)B * L * L * 4);
    float*  nrm      = (float*) alloc((size_t)ML * 4);
    bf16_t* adj_bf   = (bf16_t*)alloc((size_t)B * L * L * 2);
    bf16_t* adjti_bf = (bf16_t*)alloc((size_t)B * L * L * 2);
    float*  den      = (float*) alloc((size_t)ML * 4);
    float*  denti    = (float*) alloc((size_t)ML * 4);
    bf16_t* tmps_bf  = (bf16_t*)alloc((size_t)ML * D * 2);
    bf16_t* tmpsti_bf= (bf16_t*)alloc((size_t)ML * D * 2);
    bf16_t* t1_bf    = (bf16_t*)alloc((size_t)ML * D * 2);
    bf16_t* t2_bf    = (bf16_t*)alloc((size_t)ML * D * 2);
    float*  g1_f     = (float*) alloc((size_t)ML * D * 4);
    float*  g2_f     = (float*) alloc((size_t)ML * D * 4);
    bf16_t* cat1_bf  = (bf16_t*)alloc((size_t)B * 2 * D * 2); // [gcn_text | gcn_ti]
    bf16_t* cat2_bf  = (bf16_t*)alloc((size_t)B * 2 * D * 2); // [global_bert | local_feat]

    auto cast = [&](const float* src, bf16_t* dst, long n) {
        cast_f32_to_bf16_k<<<dim3((unsigned)((n + 255) / 256)), dim3(256), 0, stream>>>(src, dst, n);
    };
    auto castT = [&](const float* src, bf16_t* dst, int R, int C) {
        cast_transpose_k<<<dim3((unsigned)ceil_div(C, 16), (unsigned)ceil_div(R, 16)),
                           dim3(256), 0, stream>>>(src, dst, R, C);
    };
    auto gemm = [&](const bf16_t* A, const bf16_t* Bm, float* Cf, bf16_t* Cb,
                    const float* bias, const float* resid, const float* rowDiv, int relu,
                    int M, int N, int K, long lda, long ldb, long ldc,
                    int transB, float alpha, int Hn, long sAb, long sAh,
                    long sBb, long sBh, long sCb, long sCh, int batches) {
        dim3 grid((unsigned)ceil_div(N, 16 * NT), (unsigned)ceil_div(M, 16), (unsigned)batches);
        const bool full = (M % 16 == 0) && (N % (16 * NT) == 0) && (K % 32 == 0);
        if (full && transB)
            gemm_bf16_wmma<true, true><<<grid, dim3(WAVE), 0, stream>>>(
                A, Bm, Cf, Cb, bias, resid, rowDiv, relu, M, N, K, lda, ldb, ldc,
                alpha, Hn, sAb, sAh, sBb, sBh, sCb, sCh);
        else if (full)
            gemm_bf16_wmma<true, false><<<grid, dim3(WAVE), 0, stream>>>(
                A, Bm, Cf, Cb, bias, resid, rowDiv, relu, M, N, K, lda, ldb, ldc,
                alpha, Hn, sAb, sAh, sBb, sBh, sCb, sCh);
        else if (transB)
            gemm_bf16_wmma<false, true><<<grid, dim3(WAVE), 0, stream>>>(
                A, Bm, Cf, Cb, bias, resid, rowDiv, relu, M, N, K, lda, ldb, ldc,
                alpha, Hn, sAb, sAh, sBb, sBh, sCb, sCh);
        else
            gemm_bf16_wmma<false, false><<<grid, dim3(WAVE), 0, stream>>>(
                A, Bm, Cf, Cb, bias, resid, rowDiv, relu, M, N, K, lda, ldb, ldc,
                alpha, Hn, sAb, sAh, sBb, sBh, sCb, sCh);
    };

    // 1) precision casts; weights transposed so GEMMs take the vector trans-B path
    cast(vit_feature, vitbf, (long)B * NV * D);
    castT(wq, wqT, D, HD);          // -> [HD][D]
    castT(wk, wkT, D, HD);
    castT(wv, wvT, D, HD);
    castT(fc_w, fcwT, HD, D);       // -> [D][HD]
    castT(w1b, w1bT, D, D);
    castT(w2b, w2bT, D, D);
    castT(loc_w, locwT, 2 * D, D);  // -> [D][2D]
    castT(out_w, outwT, 2 * D, NC); // -> [NC][2D]

    // 2) LayerNorms (text -> f32+bf16; pooler -> bf16 straight into final concat)
    layernorm_rows_k<<<dim3(ML), dim3(256), 0, stream>>>(text_hidden, ln_g, ln_b, tf32, tfbf, D, D);
    layernorm_rows_k<<<dim3(B),  dim3(256), 0, stream>>>(pooler_out, ln_g, ln_b, nullptr, cat2_bf, 2 * D, D);

    // 3) q/k/v projections. k/v are written into (B, NVP, HD) padded blocks
    //    (batched per b) so the attention GEMMs run with padded N/K.
    gemm(tfbf, wqT, nullptr, q_bf, nullptr, nullptr, nullptr, 0,
         ML, HD, D, D, D, HD, 1, 1.f, 1, 0,0, 0,0, 0,0, 1);
    gemm(vitbf, wkT, nullptr, k_bf, nullptr, nullptr, nullptr, 0,
         NV, HD, D, D, D, HD, 1, 1.f,
         1, (long)NV * D, 0, 0, 0, (long)NVP * HD, 0, B);
    gemm(vitbf, wvT, nullptr, v_bf, nullptr, nullptr, nullptr, 0,
         NV, HD, D, D, D, HD, 1, 1.f,
         1, (long)NV * D, 0, 0, 0, (long)NVP * HD, 0, B);

    // 4) attention scores: per (b,h)  Q_h[L,768] @ K_h^T -> (L,NVP), 1/sqrt(D).
    //    N padded to 256: pad columns compute garbage into attnf pad columns
    //    (never read by softmax); all reads stay inside padded k blocks.
    const float inv_sqrt_d = 1.0f / sqrtf((float)D);
    gemm(q_bf, k_bf, attnf, nullptr, nullptr, nullptr, nullptr, 0,
         L, NVP, D, HD, HD, NVP, /*transB=*/1, inv_sqrt_d,
         H, (long)L * HD, (long)D, (long)NVP * HD, (long)D,
         (long)H * L * NVP, (long)L * NVP, BH);

    // 5) softmax over NV=197, zero-fill pad columns -> bf16 probabilities
    softmax_rows_k<<<dim3((unsigned)((long)BH * L)), dim3(WAVE), 0, stream>>>(attnf, probs_bf, NV, NVP);

    // 6) ctx: per (b,h)  P[L,NVP] @ V_h[NVP,768] (pad rows x zero probs = 0)
    //    lda=NVP is 8-aligned -> vector A loads; K=256 -> fully fast path.
    gemm(probs_bf, v_bf, nullptr, ctx_bf, nullptr, nullptr, nullptr, 0,
         L, D, NVP, NVP, HD, HD, /*transB=*/0, 1.f,
         H, (long)H * L * NVP, (long)L * NVP, (long)NVP * HD, (long)D,
         (long)L * HD, (long)D, BH);

    // 7) text_include_img = LN(ctx @ fc_w + text_feat)
    gemm(ctx_bf, fcwT, pre_f, nullptr, nullptr, /*residual=*/tf32, nullptr, 0,
         ML, D, HD, HD, HD, D, /*transB=*/1, 1.f, 1, 0,0, 0,0, 0,0, 1);
    layernorm_rows_k<<<dim3(ML), dim3(256), 0, stream>>>(pre_f, mha_ln_g, mha_ln_b, tii_f, tii_bf, D, D);

    // 8) cosine sim: dot = tii @ tii^T per b; norms from f32 tii
    rownorm_k<<<dim3(ML), dim3(256), 0, stream>>>(tii_f, nrm, D);
    gemm(tii_bf, tii_bf, dot_f, nullptr, nullptr, nullptr, nullptr, 0,
         L, L, D, D, D, L, /*transB=*/1, 1.f,
         1, (long)L * D, 0, (long)L * D, 0, (long)L * L, 0, B);
    sim_adj_k<<<dim3(L, B), dim3(256), 0, stream>>>(dot_f, nrm, adj, adj_bf, adjti_bf, den, denti, L);

    // 9) span pooling (text_feat and text_include_img)
    span_pool_k<<<dim3(L, B), dim3(256), 0, stream>>>(tf32,  tran, word_length, tmps_bf,   L, D);
    span_pool_k<<<dim3(L, B), dim3(256), 0, stream>>>(tii_f, tran, word_length, tmpsti_bf, L, D);

    // 10) GCN: t = adj @ tmps (per b), then g = relu((t @ w + b) / den)
    gemm(adj_bf, tmps_bf, nullptr, t1_bf, nullptr, nullptr, nullptr, 0,
         L, D, L, L, D, D, 0, 1.f, 1, (long)L * L, 0, (long)L * D, 0, (long)L * D, 0, B);
    gemm(adjti_bf, tmpsti_bf, nullptr, t2_bf, nullptr, nullptr, nullptr, 0,
         L, D, L, L, D, D, 0, 1.f, 1, (long)L * L, 0, (long)L * D, 0, (long)L * D, 0, B);
    gemm(t1_bf, w1bT, g1_f, nullptr, b1b, nullptr, /*rowDiv=*/den,   /*relu=*/1,
         ML, D, D, D, D, D, /*transB=*/1, 1.f, 1, 0,0, 0,0, 0,0, 1);
    gemm(t2_bf, w2bT, g2_f, nullptr, b2b, nullptr, /*rowDiv=*/denti, /*relu=*/1,
         ML, D, D, D, D, D, /*transB=*/1, 1.f, 1, 0,0, 0,0, 0,0, 1);

    // 11) masked means -> concat1 = [gcn_text | gcn_ti]
    masked_mean_k<<<dim3(B), dim3(256), 0, stream>>>(g1_f, target_mask, cat1_bf, 2 * D, 0, L, D);
    masked_mean_k<<<dim3(B), dim3(256), 0, stream>>>(g2_f, target_mask, cat1_bf, 2 * D, D, L, D);

    // 12) local_feat = concat1 @ loc_w + loc_b -> second half of concat2
    gemm(cat1_bf, locwT, nullptr, cat2_bf + D, loc_b, nullptr, nullptr, 0,
         B, D, 2 * D, 2 * D, 2 * D, 2 * D, /*transB=*/1, 1.f, 1, 0,0, 0,0, 0,0, 1);

    // 13) logits = concat2 @ out_w + out_b  (N=3 tail handled by masked loads)
    gemm(cat2_bf, outwT, out, nullptr, out_b, nullptr, nullptr, 0,
         B, NC, 2 * D, 2 * D, 2 * D, NC, /*transB=*/1, 1.f, 1, 0,0, 0,0, 0,0, 1);
}
